// NewAttention_32186484916317
// MI455X (gfx1250) — matrix operand order repaired and checked by reading, not compiled, not run on hardware
//
#include <hip/hip_runtime.h>
#include <hip/hip_bf16.h>
#include <math.h>

typedef __attribute__((ext_vector_type(16))) __bf16 v16bf;
typedef __attribute__((ext_vector_type(8)))  float  v8f;

union Frag {
    v16bf v;
    uint4 u[2];
};

#define S_LEN 2048
#define NH    16
#define DH    64
#define DMODEL 1024

// ---------------------------------------------------------------------------
// bf16 WMMA GEMM: C[4096 x 1024] = A[4096 x 1024] * W[1024 x 1024]
// 128x128 tile per 256-thread block (8 waves x 16 rows x 8 n-tiles), BK=32,
// register-prefetch pipeline (global loads for k+1 overlap WMMAs of k).
// MODE 0: out = Q bf16 [B,H,S,dh]   MODE 1: out = K bf16 [B,H,S,dh]
// MODE 2: out = V^T bf16 [B,H,dh,S] MODE 3: out = f32 [4096,1024], A is bf16
// ---------------------------------------------------------------------------
template<int MODE, bool ABF16>
__global__ __launch_bounds__(256) void gemm_kernel(
    const void* __restrict__ Aptr, const float* __restrict__ W,
    void* __restrict__ Out)
{
    const int tid  = threadIdx.x;
    const int wave = tid >> 5;
    const int lane = tid & 31;
    const int half = lane >> 4;
    const int l16  = lane & 15;
    const int mbase = blockIdx.x * 128;
    const int nbase = blockIdx.y * 128;

    __shared__ __align__(16) __bf16 Asm[128][32];
    __shared__ __align__(16) __bf16 Wsm[128][32];   // transposed: [n][k]

    v8f acc[8];
    v8f zero = {0.f,0.f,0.f,0.f,0.f,0.f,0.f,0.f};
#pragma unroll
    for (int j = 0; j < 8; ++j) acc[j] = zero;

    float4 apf[4];      // staged A (f32 path)
    uint4  apb[2];      // staged A (bf16 path)
    float4 wpf[4];      // staged W (two k-adjacent rows x 4 cols each pair)

    auto loadG = [&](int kb) {
        if constexpr (!ABF16) {
            const float* A = (const float*)Aptr;
#pragma unroll
            for (int r = 0; r < 4; ++r) {
                int idx = tid * 4 + r;               // 0..1023 float4s
                int row = idx >> 3, c4 = idx & 7;
                apf[r] = *(const float4*)(A + (size_t)(mbase + row) * DMODEL + kb + c4 * 4);
            }
        } else {
            const __bf16* A = (const __bf16*)Aptr;
#pragma unroll
            for (int r = 0; r < 2; ++r) {
                int idx = tid * 2 + r;               // 0..511 chunks of 8 bf16
                int row = idx >> 2, c8 = idx & 3;
                apb[r] = *(const uint4*)(A + (size_t)(mbase + row) * DMODEL + kb + c8 * 8);
            }
        }
#pragma unroll
        for (int r = 0; r < 2; ++r) {
            int idx = tid * 2 + r;                   // 0..511 (k-pair, col4)
            int kp = idx >> 5, c4 = idx & 31;        // k = 2*kp
            wpf[2*r+0] = *(const float4*)(W + (size_t)(kb + 2*kp)     * DMODEL + nbase + c4 * 4);
            wpf[2*r+1] = *(const float4*)(W + (size_t)(kb + 2*kp + 1) * DMODEL + nbase + c4 * 4);
        }
    };

    auto stageLDS = [&]() {
        if constexpr (!ABF16) {
#pragma unroll
            for (int r = 0; r < 4; ++r) {
                int idx = tid * 4 + r;
                int row = idx >> 3, c4 = idx & 7;
                union { __bf16 h[4]; uint2 u; } pa;
                pa.h[0] = (__bf16)apf[r].x; pa.h[1] = (__bf16)apf[r].y;
                pa.h[2] = (__bf16)apf[r].z; pa.h[3] = (__bf16)apf[r].w;
                *(uint2*)&Asm[row][c4 * 4] = pa.u;
            }
        } else {
#pragma unroll
            for (int r = 0; r < 2; ++r) {
                int idx = tid * 2 + r;
                int row = idx >> 2, c8 = idx & 3;
                *(uint4*)&Asm[row][c8 * 8] = apb[r];
            }
        }
#pragma unroll
        for (int r = 0; r < 2; ++r) {
            int idx = tid * 2 + r;
            int kp = idx >> 5, c4 = idx & 31;
            int k = 2 * kp;
            const float* w0 = &wpf[2*r+0].x;
            const float* w1 = &wpf[2*r+1].x;
#pragma unroll
            for (int j = 0; j < 4; ++j) {
                union { __bf16 h[2]; unsigned u; } pk;
                pk.h[0] = (__bf16)w0[j]; pk.h[1] = (__bf16)w1[j];
                *(unsigned*)&Wsm[c4 * 4 + j][k] = pk.u;   // packed (k,k+1)
            }
        }
    };

    loadG(0);
    for (int kb = 0; kb < DMODEL; kb += 32) {
        stageLDS();
        __syncthreads();
        if (kb + 32 < DMODEL) loadG(kb + 32);   // overlaps with WMMAs below

        Frag af;
        const __bf16* arow = &Asm[wave * 16 + l16][0];
        af.u[0] = *(const uint4*)(arow + (half ? 16 : 0));
        af.u[1] = *(const uint4*)(arow + (half ? 24 : 8));
#pragma unroll
        for (int j = 0; j < 8; ++j) {
            Frag bfr;
            const __bf16* wrow = &Wsm[j * 16 + l16][0] + (half ? 16 : 0);
            bfr.u[0] = *(const uint4*)(wrow);
            bfr.u[1] = *(const uint4*)(wrow + 8);
            acc[j] = __builtin_amdgcn_wmma_f32_16x16x32_bf16(
                false, af.v, false, bfr.v, (short)0, acc[j], false, false);
        }
        __syncthreads();
    }

    // ---- epilogue ----
#pragma unroll
    for (int j = 0; j < 8; ++j) {
#pragma unroll
        for (int i = 0; i < 8; ++i) {
            float v = acc[j][i];
            int row = mbase + wave * 16 + i + (half ? 8 : 0);
            int col = nbase + j * 16 + l16;
            if constexpr (MODE == 3) {
                ((float*)Out)[(size_t)row * DMODEL + col] = v;
            } else {
                int b = row >> 11, s = row & (S_LEN - 1);
                int h = col >> 6,  d = col & (DH - 1);
                size_t o;
                if constexpr (MODE == 2)
                    o = ((size_t)(b * NH + h) * DH + d) * S_LEN + s;   // V^T
                else
                    o = ((size_t)(b * NH + h) * S_LEN + s) * DH + d;   // Q, K
                ((__bf16*)Out)[o] = (__bf16)v;
            }
        }
    }
}

// ---------------------------------------------------------------------------
// RoPE in-place on bf16 [B*H, S, DH]; one thread per (row, pair)
// ---------------------------------------------------------------------------
__global__ __launch_bounds__(256) void rope_kernel(__bf16* __restrict__ T)
{
    size_t idx = (size_t)blockIdx.x * blockDim.x + threadIdx.x;  // B*H*S*32
    int i = (int)(idx & 31);            // pair index 0..31
    size_t rowp = idx >> 5;             // (bh, s)
    int s = (int)(rowp & (S_LEN - 1));
    float freq = expf(-(float)(2 * i) * (1.0f / 64.0f) * 9.210340371976184f); // ln(10000)
    float ang = (float)s * freq;
    float c = cosf(ang), sn = sinf(ang);
    __bf16* p = T + rowp * DH + 2 * i;
    float a = (float)p[0], b = (float)p[1];
    p[0] = (__bf16)(a * c - b * sn);
    p[1] = (__bf16)(a * sn + b * c);
}

// ---------------------------------------------------------------------------
// Causal flash attention. grid = (S/64, B*H), block = 128 (4 waves).
// Wave w handles 16 query rows. Q,K: [B,H,S,dh] bf16; Vt: [B,H,dh,S] bf16.
// O: [B,S,H*dh] bf16. Row sums of P are accumulated with an all-ones B
// fragment WMMA (C[m][n] = sum_k P[m][k] for every n), so no sum butterfly
// and no final broadcast. Plds is wave-private: DS is in-order per wave.
// ---------------------------------------------------------------------------
__global__ __launch_bounds__(128) void attn_kernel(
    const __bf16* __restrict__ Q, const __bf16* __restrict__ K,
    const __bf16* __restrict__ Vt, __bf16* __restrict__ O)
{
    const int bh   = blockIdx.y;                 // b*16 + h
    const int qblk = blockIdx.x;                 // 64 query rows
    const int tid  = threadIdx.x;
    const int wave = tid >> 5;
    const int lane = tid & 31;
    const int half = lane >> 4;
    const int l16  = lane & 15;

    const __bf16* Qb = Q  + (size_t)bh * S_LEN * DH;
    const __bf16* Kb = K  + (size_t)bh * S_LEN * DH;
    const __bf16* Vb = Vt + (size_t)bh * DH * S_LEN;

    const int qw = qblk * 64 + wave * 16;        // wave's first query row

    __shared__ __align__(16) __bf16 Plds[4][16][32];

    // Q fragments for dh k-steps 0 and 32 (A-layout 16x32)
    Frag qf[2];
    {
        const __bf16* qrow = Qb + (size_t)(qw + l16) * DH;
#pragma unroll
        for (int ks = 0; ks < 2; ++ks) {
            qf[ks].u[0] = *(const uint4*)(qrow + ks * 32 + (half ? 16 : 0));
            qf[ks].u[1] = *(const uint4*)(qrow + ks * 32 + (half ? 24 : 8));
        }
    }

    // all-ones B fragment for row-sum accumulation
    Frag ones;
#pragma unroll
    for (int e = 0; e < 16; ++e) ones.v[e] = (__bf16)1.0f;

    v8f zero = {0.f,0.f,0.f,0.f,0.f,0.f,0.f,0.f};
    v8f oacc[4], lacc = zero;
#pragma unroll
    for (int d4 = 0; d4 < 4; ++d4) oacc[d4] = zero;
    float m_st[8];
#pragma unroll
    for (int i = 0; i < 8; ++i) m_st[i] = -INFINITY;

    const int kmax = qblk * 64 + 64;             // workgroup upper bound
    for (int kb = 0; kb < kmax; kb += 32) {
        if (kb > qw + 15) continue;              // fully masked for this wave

        // prefetch next K/V block (global_prefetch_b8; cacheline per lane)
        if (kb + 32 <= qw + 15) {
            __builtin_prefetch(Kb + (size_t)(kb + 32 + lane) * DH, 0, 3);
            __builtin_prefetch(Vb + (size_t)lane        * S_LEN + kb + 32, 0, 3);
            __builtin_prefetch(Vb + (size_t)(lane + 32) * S_LEN + kb + 32, 0, 3);
        }

        // ---- load all K and V fragments for this block up front ----
        Frag kf[2][2], vf[4];
#pragma unroll
        for (int j = 0; j < 2; ++j) {
            const __bf16* krow = Kb + (size_t)(kb + j * 16 + l16) * DH;
            const int koff = half ? 16 : 0;
            kf[j][0].u[0] = *(const uint4*)(krow + koff);
            kf[j][0].u[1] = *(const uint4*)(krow + koff + 8);
            kf[j][1].u[0] = *(const uint4*)(krow + 32 + koff);
            kf[j][1].u[1] = *(const uint4*)(krow + 32 + koff + 8);
        }
#pragma unroll
        for (int d4 = 0; d4 < 4; ++d4) {
            const __bf16* vrow = Vb + (size_t)(d4 * 16 + l16) * S_LEN + kb + (half ? 16 : 0);
            vf[d4].u[0] = *(const uint4*)(vrow);
            vf[d4].u[1] = *(const uint4*)(vrow + 8);
        }

        // ---- S = Q K^T for two 16-key tiles ----
        v8f sc[2];
#pragma unroll
        for (int j = 0; j < 2; ++j) {
            v8f c = zero;
            c = __builtin_amdgcn_wmma_f32_16x16x32_bf16(
                false, qf[0].v, false, kf[j][0].v, (short)0, c, false, false);
            c = __builtin_amdgcn_wmma_f32_16x16x32_bf16(
                false, qf[1].v, false, kf[j][1].v, (short)0, c, false, false);
            sc[j] = c;
        }

        // ---- scale + causal mask (mask path only near the diagonal) ----
        float p[2][8];
        if (kb + 31 > qw) {
#pragma unroll
            for (int j = 0; j < 2; ++j) {
                int key = kb + j * 16 + l16;
#pragma unroll
                for (int i = 0; i < 8; ++i) {
                    int q = qw + i + (half ? 8 : 0);
                    float s = sc[j][i] * 0.125f;          // 1/sqrt(64)
                    p[j][i] = (key > q) ? -INFINITY : s;
                }
            }
        } else {
#pragma unroll
            for (int j = 0; j < 2; ++j)
#pragma unroll
                for (int i = 0; i < 8; ++i)
                    p[j][i] = sc[j][i] * 0.125f;
        }

        // ---- online softmax: max butterfly + exp + alpha rescale ----
#pragma unroll
        for (int i = 0; i < 8; ++i) {
            float mx = fmaxf(p[0][i], p[1][i]);
#pragma unroll
            for (int off = 1; off < 16; off <<= 1)
                mx = fmaxf(mx, __shfl_xor(mx, off, 32));
            float mcur = fmaxf(m_st[i], mx);
            float alpha = __expf(m_st[i] - mcur);
            p[0][i] = __expf(p[0][i] - mcur);
            p[1][i] = __expf(p[1][i] - mcur);
            m_st[i] = mcur;
            lacc[i] *= alpha;
#pragma unroll
            for (int d4 = 0; d4 < 4; ++d4) oacc[d4][i] *= alpha;
        }

        // ---- P (C-layout) -> LDS -> A-layout fragment (wave-private) ----
#pragma unroll
        for (int j = 0; j < 2; ++j)
#pragma unroll
            for (int i = 0; i < 8; ++i)
                Plds[wave][i + (half ? 8 : 0)][j * 16 + l16] = (__bf16)p[j][i];
        __builtin_amdgcn_wave_barrier();   // codegen fence; DS is in-order per wave

        Frag pf;
        const __bf16* prow = &Plds[wave][l16][0];
        pf.u[0] = *(const uint4*)(prow + (half ? 16 : 0));
        pf.u[1] = *(const uint4*)(prow + (half ? 24 : 8));

        // ---- row sums and O += P V ----
        lacc = __builtin_amdgcn_wmma_f32_16x16x32_bf16(
            false, pf.v, false, ones.v, (short)0, lacc, false, false);
#pragma unroll
        for (int d4 = 0; d4 < 4; ++d4) {
            oacc[d4] = __builtin_amdgcn_wmma_f32_16x16x32_bf16(
                false, pf.v, false, vf[d4].v, (short)0, oacc[d4], false, false);
        }
        __builtin_amdgcn_wave_barrier();
    }

    // ---- normalize and write out [B,S,H*dh] ----
    const int b = bh >> 4, h = bh & 15;
#pragma unroll
    for (int d4 = 0; d4 < 4; ++d4)
#pragma unroll
        for (int i = 0; i < 8; ++i) {
            int s = qw + i + (half ? 8 : 0);
            int col = h * DH + d4 * 16 + l16;
            float v = oacc[d4][i] / lacc[i];
            O[((size_t)(b * S_LEN + s)) * DMODEL + col] = (__bf16)v;
        }
}

// ---------------------------------------------------------------------------
extern "C" void kernel_launch(void* const* d_in, const int* in_sizes, int n_in,
                              void* d_out, int out_size, void* d_ws, size_t ws_size,
                              hipStream_t stream)
{
    const float* x  = (const float*)d_in[0];
    const float* wq = (const float*)d_in[1];
    const float* wk = (const float*)d_in[2];
    const float* wv = (const float*)d_in[3];
    const float* wo = (const float*)d_in[4];

    char* ws = (char*)d_ws;
    const size_t seg = (size_t)2 * NH * S_LEN * DH * sizeof(__bf16);   // 8 MB
    __bf16* Qb = (__bf16*)(ws);
    __bf16* Kb = (__bf16*)(ws + seg);
    __bf16* Vt = (__bf16*)(ws + 2 * seg);
    __bf16* AO = (__bf16*)(ws + 3 * seg);

    dim3 gg(32, 8), bg(256);
    gemm_kernel<0, false><<<gg, bg, 0, stream>>>(x, wq, Qb);
    gemm_kernel<1, false><<<gg, bg, 0, stream>>>(x, wk, Kb);
    gemm_kernel<2, false><<<gg, bg, 0, stream>>>(x, wv, Vt);

    const int pairs = 2 * NH * S_LEN * (DH / 2);   // 2,097,152
    rope_kernel<<<pairs / 256, 256, 0, stream>>>(Qb);
    rope_kernel<<<pairs / 256, 256, 0, stream>>>(Kb);

    attn_kernel<<<dim3(S_LEN / 64, 2 * NH), 128, 0, stream>>>(Qb, Kb, Vt, AO);

    gemm_kernel<3, true><<<gg, bg, 0, stream>>>(AO, wo, d_out);
}
